// MoeLayer_77335181132293
// MI455X (gfx1250) — matrix operand order repaired and checked by reading, not compiled, not run on hardware
//
#include <hip/hip_runtime.h>
#include <hip/hip_bf16.h>

// ---------------- problem constants (from reference) ----------------
#define TOK   4096      // B*S
#define DIM   1024      // D
#define NEXP  16        // E
#define HID   4096      // H
#define CAP   320       // int(1.25 * T / E)

typedef __attribute__((ext_vector_type(16))) __bf16     v16bf;
typedef __attribute__((ext_vector_type(8)))  float      v8f;
typedef __attribute__((ext_vector_type(8)))  unsigned   v8u;
typedef __attribute__((ext_vector_type(4)))  unsigned   v4u;

__device__ __forceinline__ unsigned short bf16_of(float f) {
    unsigned u = __builtin_bit_cast(unsigned, f);
    u = (u + 0x7FFFu + ((u >> 16) & 1u)) >> 16;   // round-to-nearest-even
    return (unsigned short)u;
}
__device__ __forceinline__ unsigned pk_bf16(float lo, float hi) {
    unsigned ua = __builtin_bit_cast(unsigned, lo);
    unsigned ub = __builtin_bit_cast(unsigned, hi);
    ua = (ua + 0x7FFFu + ((ua >> 16) & 1u)) >> 16;
    ub = (ub + 0x7FFFu + ((ub >> 16) & 1u)) >> 16;
    return (ub << 16) | (ua & 0xFFFFu);
}

// ---------------- 1) gating: logits -> argmax + softmax weight ----------------
// one wave per token; lane = (expert = lane&15, half = lane>>4)
__global__ void gate_kernel(const float* __restrict__ tokens,
                            const float* __restrict__ gate,
                            int* __restrict__ eidx, float* __restrict__ wgt) {
    const int lane = threadIdx.x & 31;
    const int t    = (blockIdx.x * blockDim.x + threadIdx.x) >> 5;
    if (t >= TOK) return;
    const int e = lane & 15, half = lane >> 4;

    const float4* tk = (const float4*)(tokens + (size_t)t * DIM + half * (DIM / 2));
    const float4* gw = (const float4*)(gate   + (size_t)e * DIM + half * (DIM / 2));
    float acc = 0.f;
    #pragma unroll 4
    for (int i = 0; i < DIM / 8; ++i) {
        float4 a = tk[i], b = gw[i];
        acc += a.x * b.x + a.y * b.y + a.z * b.z + a.w * b.w;
    }
    acc += __shfl_xor(acc, 16, 32);            // lanes 0-15 now hold logits[e]

    float lg = (lane < 16) ? acc : -3.4e38f;
    int   ie = (lane < 16) ? e   : 0x7fffffff;
    float ml = lg; int mi = ie;
    #pragma unroll
    for (int off = 16; off; off >>= 1) {       // max + first-index argmax
        float ol = __shfl_xor(ml, off, 32);
        int   oi = __shfl_xor(mi, off, 32);
        if (ol > ml || (ol == ml && oi < mi)) { ml = ol; mi = oi; }
    }
    float ex = (lane < 16) ? __expf(lg - ml) : 0.f;
    #pragma unroll
    for (int off = 16; off; off >>= 1) ex += __shfl_xor(ex, off, 32);
    if (lane == 0) { eidx[t] = mi; wgt[t] = 1.f / ex; }  // prob of argmax
}

// ---------------- 2) order-preserving slot assignment ----------------
__global__ void slot_kernel(const int* __restrict__ eidx,
                            int* __restrict__ slot_of, int* __restrict__ tok_of) {
    const int e = threadIdx.x;
    if (e >= NEXP) return;
    for (int c = 0; c < CAP; ++c) tok_of[e * CAP + c] = -1;
    int cnt = 0;
    for (int t = 0; t < TOK; ++t) {
        if (eidx[t] == e) {
            if (cnt < CAP) { slot_of[t] = cnt; tok_of[e * CAP + cnt] = t; }
            else           { slot_of[t] = -1; }
            ++cnt;
        }
    }
}

// ---------------- 3) dispatch gather -> bf16 [E][C][D] ----------------
__global__ void dispatch_kernel(const float* __restrict__ tokens,
                                const int* __restrict__ tok_of,
                                unsigned short* __restrict__ X) {
    const int row = blockIdx.x;                // e*CAP + c
    const int t   = tok_of[row];
    for (int d = threadIdx.x; d < DIM; d += blockDim.x) {
        float v = (t >= 0) ? tokens[(size_t)t * DIM + d] : 0.f;
        X[(size_t)row * DIM + d] = bf16_of(v);
    }
}

// ---------------- 4/5) expert GEMM, double-buffered LDS B panel ----------------
// Block = 128 threads = 4 waves; block computes a 64(M) x 64(N) tile.
// A: bf16 [E][CAP][K] row-major (two b128 global loads per lane = ISA A layout).
// B: f32 [E][K][N] row-major -> coalesced float4 loads (pipelined one K-step
//    ahead in registers) -> bf16 K-pair dwords in LDS [n][kpair], row stride
//    20 dwords (16B-aligned, bank-conflict-free). Double-buffered: ONE barrier
//    per K-step. Consumer reads each 16x16 B fragment as 2 x ds_load_b128,
//    all 8 fragment loads issued before the 4 back-to-back WMMAs.
template <bool RELU, bool OUT_BF16, int K, int N>
__global__ __launch_bounds__(128)
void ffn_gemm(const unsigned short* __restrict__ Abase,
              const float* __restrict__ Bbase,
              void* __restrict__ Obase) {
    __shared__ unsigned lsB[2 * 64 * 20];      // 2 x 5 KB B panels

    const int tid  = threadIdx.x;
    const int lane = tid & 31;
    const int wave = tid >> 5;                 // M sub-tile within block
    const int e  = blockIdx.z;
    const int c0 = blockIdx.y * 64 + wave * 16;
    const int n0 = blockIdx.x * 64;
    const int hi = lane >> 4;
    const int ln = lane & 15;

    // producer mapping: unit u in {0,1}: kpair = tid/16 + 8u, n4 = (tid%16)*4
    const int pn4  = (tid & 15) * 4;
    const int pkp0 = tid >> 4;

    const unsigned short* Arow = Abase + ((size_t)(e * CAP + c0 + ln)) * K;
    const float*          Bpan = Bbase + (size_t)e * K * N + n0 + pn4;

    const int      kA   = hi * 8;                             // A lane K offset
    const unsigned ldsC = (unsigned)((ln * 20 + hi * 4) * 4); // consumer byte base

    // -------- prologue: first panel into registers --------
    float4 r[2][2];
    #pragma unroll
    for (int u = 0; u < 2; ++u) {
        const float* bp = Bpan + (size_t)(2 * (pkp0 + 8 * u)) * N;
        r[u][0] = *(const float4*)bp;          // row k   (even)
        r[u][1] = *(const float4*)(bp + N);    // row k+1 (odd)
    }

    v8f acc[4] = {};
    unsigned buf = 0;
    for (int k0 = 0; k0 < K; k0 += 32) {
        // ---- producer: cvt + store panel k0 into LDS buf ----
        unsigned* L = lsB + buf * (64 * 20);
        #pragma unroll
        for (int u = 0; u < 2; ++u) {
            const int kp = pkp0 + 8 * u;
            L[(pn4 + 0) * 20 + kp] = pk_bf16(r[u][0].x, r[u][1].x);
            L[(pn4 + 1) * 20 + kp] = pk_bf16(r[u][0].y, r[u][1].y);
            L[(pn4 + 2) * 20 + kp] = pk_bf16(r[u][0].z, r[u][1].z);
            L[(pn4 + 3) * 20 + kp] = pk_bf16(r[u][0].w, r[u][1].w);
        }
        __syncthreads();                       // panel visible; prev buf consumed

        // ---- pipeline: issue global loads for panel k0+32 ----
        if (k0 + 32 < K) {
            #pragma unroll
            for (int u = 0; u < 2; ++u) {
                const float* bp = Bpan + (size_t)(k0 + 32 + 2 * (pkp0 + 8 * u)) * N;
                r[u][0] = *(const float4*)bp;
                r[u][1] = *(const float4*)(bp + N);
            }
            if (k0 + 64 < K)                   // warm L2 two panels ahead
                __builtin_prefetch(Bpan + (size_t)(k0 + 64 + 2 * pkp0) * N, 0, 1);
        }

        // ---- consumer: A tile (2 x b128 global) + 4 B fragments (8 x ds_load_b128) ----
        v4u alo = *(const v4u*)(Arow + k0 + kA);
        v4u ahi = *(const v4u*)(Arow + k0 + kA + 16);
        v8u apk = {alo[0], alo[1], alo[2], alo[3], ahi[0], ahi[1], ahi[2], ahi[3]};
        v16bf A = __builtin_bit_cast(v16bf, apk);

        const char* bl = (const char*)(lsB + buf * (64 * 20)) + ldsC;
        v4u bf[8];
        #pragma unroll
        for (int tle = 0; tle < 4; ++tle) {
            bf[2 * tle + 0] = *(const v4u*)(bl + tle * (16 * 20 * 4));
            bf[2 * tle + 1] = *(const v4u*)(bl + tle * (16 * 20 * 4) + 32);
        }
        #pragma unroll
        for (int tle = 0; tle < 4; ++tle) {
            v4u b0 = bf[2 * tle], b1 = bf[2 * tle + 1];
            v8u bpk = {b0[0], b0[1], b0[2], b0[3], b1[0], b1[1], b1[2], b1[3]};
            v16bf Bv = __builtin_bit_cast(v16bf, bpk);
            acc[tle] = __builtin_amdgcn_wmma_f32_16x16x32_bf16(
                false, A, false, Bv, (short)0, acc[tle], false, false);
        }
        buf ^= 1;
    }

    // epilogue: D layout -> lane<16: (row r, n=lane); lane>=16: (row r+8, n=lane-16)
    #pragma unroll
    for (int tle = 0; tle < 4; ++tle) {
        #pragma unroll
        for (int rr = 0; rr < 8; ++rr) {
            const int mm = rr + hi * 8;
            float v = acc[tle][rr];
            if (RELU) v = v > 0.f ? v : 0.f;
            const size_t idx = ((size_t)(e * CAP + c0 + mm)) * N + n0 + tle * 16 + ln;
            if (OUT_BF16) ((unsigned short*)Obase)[idx] = bf16_of(v);
            else          ((float*)Obase)[idx] = v;
        }
    }
}

// ---------------- 6) combine / scatter ----------------
__global__ void combine_kernel(const float* __restrict__ Y,
                               const int* __restrict__ eidx,
                               const int* __restrict__ slot_of,
                               const float* __restrict__ wgt,
                               float* __restrict__ out) {
    const int t = blockIdx.x;
    const int e = eidx[t], s = slot_of[t];
    const float w = wgt[t];
    if (s >= 0) {
        const float* yr = Y + ((size_t)(e * CAP + s)) * DIM;
        for (int d = threadIdx.x; d < DIM; d += blockDim.x)
            out[(size_t)t * DIM + d] = w * yr[d];
    } else {
        for (int d = threadIdx.x; d < DIM; d += blockDim.x)
            out[(size_t)t * DIM + d] = 0.f;
    }
}

// ---------------- launcher ----------------
extern "C" void kernel_launch(void* const* d_in, const int* in_sizes, int n_in,
                              void* d_out, int out_size, void* d_ws, size_t ws_size,
                              hipStream_t stream) {
    const float* tokens = (const float*)d_in[0];   // [T, D]
    const float* gate   = (const float*)d_in[1];   // [E, D]
    const float* w1     = (const float*)d_in[2];   // [E, D, H]
    const float* w2     = (const float*)d_in[3];   // [E, H, D]
    float* out = (float*)d_out;

    char* ws = (char*)d_ws;
    int*            eidx    = (int*)(ws + 0);            // T ints
    int*            slot_of = (int*)(ws + 16384);        // T ints
    float*          wgt     = (float*)(ws + 32768);      // T floats
    int*            tok_of  = (int*)(ws + 49152);        // E*CAP ints
    unsigned short* Xd      = (unsigned short*)(ws + 69632);     // [E][C][D] bf16 (10 MB)
    unsigned short* Hb      = (unsigned short*)(ws + 10555392);  // [E][C][H] bf16 (40 MB)
    float*          Yb      = (float*)(ws + 52498432);           // [E][C][D] f32  (20 MB)

    gate_kernel<<<TOK / 8, 256, 0, stream>>>(tokens, gate, eidx, wgt);
    slot_kernel<<<1, 32, 0, stream>>>(eidx, slot_of, tok_of);
    dispatch_kernel<<<NEXP * CAP, 256, 0, stream>>>(tokens, tok_of, Xd);
    ffn_gemm<true,  true,  DIM, HID><<<dim3(HID / 64, CAP / 64, NEXP), 128, 0, stream>>>(Xd, w1, Hb);
    ffn_gemm<false, false, HID, DIM><<<dim3(DIM / 64, CAP / 64, NEXP), 128, 0, stream>>>(Hb, w2, Yb);
    combine_kernel<<<TOK, 256, 0, stream>>>(Yb, eidx, slot_of, wgt, out);
}
